// Density_prop_with_pad_UNET_55379308314942
// MI455X (gfx1250) — matrix operand order repaired and checked by reading, not compiled, not run on hardware
//
#include <hip/hip_runtime.h>
#include <hip/hip_bf16.h>
#include <cstdint>

// CDNA5 / gfx1250 wave32 WMMA types
typedef __attribute__((ext_vector_type(16))) _Float16 v16h;
typedef __attribute__((ext_vector_type(8)))  float    v8f;

union FragA { v16h v; _Float16 h[16]; };
union FragC { v8f  v; float     f[8];  };

#define TMB 64    // output pixels per block (4 sub-tiles of 16)
#define MT  4
#define KSTEP 32  // K per WMMA step
#define TN 64     // couts per block (4 waves x 16)

__device__ __forceinline__ float softplus_f(float x) {
    return logf(1.0f + expf(x));
}

// low 32 bits of a flat LDS pointer == wave-relative LDS byte offset (ISA 10.2)
__device__ __forceinline__ unsigned lds_off_of(const void* p) {
    return (unsigned)(uintptr_t)p;
}

// ---------------------------------------------------------------------------
// One-time weight prep: f32 -> f16 tables of w and w^2 (HWIO flattened [K][Cout])
// ---------------------------------------------------------------------------
__global__ void wprep_kernel(const float* __restrict__ w,
                             _Float16* __restrict__ wh,
                             _Float16* __restrict__ wh2, int n)
{
    int i = (int)blockIdx.x * blockDim.x + threadIdx.x;
    if (i >= n) return;
    float v = w[i];
    wh[i]  = (_Float16)v;
    wh2[i] = (_Float16)(v * v);
}

// ---------------------------------------------------------------------------
// Fused density-propagating conv layer, implicit GEMM via WMMA.
//   mu_out = conv(mu_in, w_mu)                         [f16 WMMA, f32 acc]
//   s_out  = patchsum(mu_in^2 + s_in) * softplus(w_sig) + conv(s_in, w_mu^2)
//   optional ReLU gating.
// Tile: M=64 pixels x N=64 couts, 4 waves; each wave: 1 B fragment reused over
// 4 A sub-tiles -> 8 WMMAs/step. B tiles staged via global_load_async_to_lds.
// ---------------------------------------------------------------------------
__global__ __launch_bounds__(128) void conv_wmma_kernel(
    const float* __restrict__ mu_in, const float* __restrict__ s_in,
    const _Float16* __restrict__ wh, const _Float16* __restrict__ wh2,
    const float* __restrict__ w_sig,
    float* __restrict__ mu_out, float* __restrict__ s_out,
    int Bn, int Hin, int Win, int Cin, int Cout, int Ho, int Wo,
    int pad, int relu)
{
    __shared__ __align__(16) _Float16 Amu[TMB][KSTEP];
    __shared__ __align__(16) _Float16 As [TMB][KSTEP];
    __shared__ __align__(16) _Float16 Bw [KSTEP][TN];
    __shared__ __align__(16) _Float16 Bw2[KSTEP][TN];
    __shared__ float psum[TMB];
    __shared__ int   pb[TMB], piy[TMB], pjx[TMB], pval[TMB];

    const int tid  = threadIdx.x;
    const int lane = tid & 31;
    const int wave = tid >> 5;
    const int npix = Bn * Ho * Wo;
    const int pix0 = (int)blockIdx.x * TMB;
    const int cout0 = (int)blockIdx.y * TN;
    const int Ktot = 9 * Cin;
    const int ksteps = (Ktot + KSTEP - 1) / KSTEP;
    const bool has_s = (s_in != nullptr);

    // cache per-pixel decomposition once
    if (tid < TMB) {
        int gp = pix0 + tid;
        int v = gp < npix;
        int b = 0, oy = 0, ox = 0;
        if (v) {
            b = gp / (Ho * Wo);
            int r = gp % (Ho * Wo);
            oy = r / Wo; ox = r % Wo;
        }
        pb[tid] = b; piy[tid] = oy - pad; pjx[tid] = ox - pad;
        pval[tid] = v;
        psum[tid] = 0.0f;
    }
    __syncthreads();

    const unsigned ldsBw  = lds_off_of(&Bw[0][0]);
    const unsigned ldsBw2 = lds_off_of(&Bw2[0][0]);

    FragC accm[MT], accs[MT];
    #pragma unroll
    for (int mt = 0; mt < MT; ++mt) {
        accm[mt].v = (v8f){0,0,0,0,0,0,0,0};
        accs[mt].v = (v8f){0,0,0,0,0,0,0,0};
    }

    for (int ks = 0; ks < ksteps; ++ks) {
        const int k0 = ks * KSTEP;

        // ---- B tiles: 32 K-rows x 64 couts of f16 (128B rows), async -> LDS
        #pragma unroll
        for (int j = 0; j < 2; ++j) {
            int i  = tid + j * 128;          // 16B chunk id, 0..255
            int kk = i >> 3, ch = i & 7;
            int k  = k0 + kk;
            unsigned loff = (unsigned)(kk * (TN * 2) + ch * 16);
            if (k < Ktot) {
                unsigned goff = (unsigned)(((long)k * Cout + cout0) * 2 + ch * 16);
                asm volatile("global_load_async_to_lds_b128 %0, %1, %2"
                             :: "v"(ldsBw + loff), "v"(goff), "s"(wh) : "memory");
                if (has_s)
                    asm volatile("global_load_async_to_lds_b128 %0, %1, %2"
                                 :: "v"(ldsBw2 + loff), "v"(goff), "s"(wh2) : "memory");
            } else {                          // zero-fill ragged K (Cin=3 layer)
                #pragma unroll
                for (int t = 0; t < 8; ++t) {
                    Bw[kk][ch * 8 + t] = (_Float16)0.0f;
                    Bw2[kk][ch * 8 + t] = (_Float16)0.0f;
                }
            }
        }

        // ---- A tiles: im2col gather, fixed k-column per thread
        const int kk = tid & 31;
        const int k  = k0 + kk;
        const int kvalid = (k < Ktot);
        int cin = 0, ky = 0, kx = 0;
        if (kvalid) {
            cin = k % Cin;
            int cell = k / Cin;               // ky*3+kx (HWIO)
            ky = cell / 3; kx = cell % 3;
        }
        for (int j = 0; j < 16; ++j) {
            int p = wave + 4 * j;             // waves own disjoint pixel rows
            float vm = 0.0f, vs = 0.0f;
            if (kvalid && pval[p]) {
                int iy = piy[p] + ky, ix = pjx[p] + kx;
                if (iy >= 0 && iy < Hin && ix >= 0 && ix < Win) {
                    long idx = (((long)pb[p] * Hin + iy) * Win + ix) * Cin + cin;
                    vm = mu_in[idx];
                    if (has_s) vs = s_in[idx];
                }
            }
            Amu[p][kk] = (_Float16)vm;
            if (has_s) As[p][kk] = (_Float16)vs;
            // patch power sum: whole wave holds one pixel row -> shuffle reduce
            float val = vm * vm + vs;
            #pragma unroll
            for (int off = 16; off > 0; off >>= 1)
                val += __shfl_xor(val, off, 32);
            if (lane == 0) psum[p] += val;    // exclusive per wave, no atomic
        }

        asm volatile("s_wait_asynccnt 0" ::: "memory");
        __syncthreads();

        // ---- fragments + WMMA
        // B 32x16 f16: col = lane&15, K halves {kb..kb+7,16+kb..16+kb+7}, kb=(lane>>4)*8
        const int cc   = wave * 16 + (lane & 15);
        const int koff = (lane >> 4) * 8;
        FragA bw, bw2;
        #pragma unroll
        for (int h = 0; h < 16; ++h) bw.h[h] = Bw[koff + (h < 8 ? h : h + 8)][cc];
        if (has_s) {
            #pragma unroll
            for (int h = 0; h < 16; ++h) bw2.h[h] = Bw2[koff + (h < 8 ? h : h + 8)][cc];
        }
        // A 16x32 f16: row=lane&15 (+subtile), K halves {kb..kb+7,16+kb..16+kb+7}
        const int kbase = (lane >> 4) * 8;
        #pragma unroll
        for (int mt = 0; mt < MT; ++mt) {
            const int row = mt * 16 + (lane & 15);
            FragA am;
            #pragma unroll
            for (int i = 0; i < 16; ++i) {
                int K = (i < 8) ? (kbase + i) : (16 + kbase + (i - 8));
                am.h[i] = Amu[row][K];
            }
            accm[mt].v = __builtin_amdgcn_wmma_f32_16x16x32_f16(
                false, am.v, false, bw.v, (short)0, accm[mt].v, false, false);
            if (has_s) {
                FragA asv;
                #pragma unroll
                for (int i = 0; i < 16; ++i) {
                    int K = (i < 8) ? (kbase + i) : (16 + kbase + (i - 8));
                    asv.h[i] = As[row][K];
                }
                accs[mt].v = __builtin_amdgcn_wmma_f32_16x16x32_f16(
                    false, asv.v, false, bw2.v, (short)0, accs[mt].v, false, false);
            }
        }
        __syncthreads();
    }

    // ---- epilogue: C/D layout lane l, vgpr r -> M = r + 8*(l>>4), N = l&15
    const int cout = cout0 + wave * 16 + (lane & 15);
    float wsv = (cout < Cout) ? softplus_f(w_sig[cout]) : 0.0f;
    #pragma unroll
    for (int mt = 0; mt < MT; ++mt) {
        #pragma unroll
        for (int r = 0; r < 8; ++r) {
            int m  = mt * 16 + r + 8 * (lane >> 4);
            int gp = pix0 + m;
            if (gp >= npix || cout >= Cout) continue;
            float mu = accm[mt].f[r];
            float sg = psum[m] * wsv + (has_s ? accs[mt].f[r] : 0.0f);
            if (relu && mu <= 0.0f) { mu = 0.0f; sg = 0.0f; }
            long o = (long)gp * Cout + cout;
            mu_out[o] = mu;
            s_out[o]  = sg;
        }
    }
}

// ---------------------------------------------------------------------------
// 2x2/stride-2 argmax pool; sigma gathered at argmax (first-max tie break).
// ---------------------------------------------------------------------------
__global__ void pool_kernel(const float* __restrict__ mu_in,
                            const float* __restrict__ s_in,
                            float* __restrict__ mu_out, float* __restrict__ s_out,
                            int Bn, int H, int W, int C)
{
    int Ho = H / 2, Wo = W / 2;
    long nidx = (long)blockIdx.x * blockDim.x + threadIdx.x;
    long tot = (long)Bn * Ho * Wo * C;
    if (nidx >= tot) return;
    int c = (int)(nidx % C); long r = nidx / C;
    int ox = (int)(r % Wo); r /= Wo;
    int oy = (int)(r % Ho); int b = (int)(r / Ho);
    float best = -3.402823466e38f, bs = 0.0f;
    #pragma unroll
    for (int e = 0; e < 4; ++e) {
        int dy = e >> 1, dx = e & 1;
        long idx = (((long)b * H + (2 * oy + dy)) * W + (2 * ox + dx)) * C + c;
        float m = mu_in[idx];
        if (m > best) { best = m; bs = s_in[idx]; }
    }
    mu_out[nidx] = best;
    s_out[nidx]  = bs;
}

// ---------------------------------------------------------------------------
// Zero-insertion unpool: out[b,2i+1,2j+1,c] = in[b,i,j,c]; out [B,2H+1,2W+1,C]
// ---------------------------------------------------------------------------
__global__ void unpool_kernel(const float* __restrict__ mu_in,
                              const float* __restrict__ s_in,
                              float* __restrict__ mu_out, float* __restrict__ s_out,
                              int Bn, int H, int W, int C)
{
    int Ho = 2 * H + 1, Wo = 2 * W + 1;
    long nidx = (long)blockIdx.x * blockDim.x + threadIdx.x;
    long tot = (long)Bn * Ho * Wo * C;
    if (nidx >= tot) return;
    int c = (int)(nidx % C); long r = nidx / C;
    int ox = (int)(r % Wo); r /= Wo;
    int oy = (int)(r % Ho); int b = (int)(r / Ho);
    float m = 0.0f, s = 0.0f;
    if ((oy & 1) && (ox & 1)) {
        long idx = (((long)b * H + (oy >> 1)) * W + (ox >> 1)) * C + c;
        m = mu_in[idx]; s = s_in[idx];
    }
    mu_out[nidx] = m;
    s_out[nidx]  = s;
}

// ---------------------------------------------------------------------------
// Channel concat with center-crop of encoder skip.
// ---------------------------------------------------------------------------
__global__ void concat_kernel(const float* __restrict__ dmu, const float* __restrict__ dsg,
                              const float* __restrict__ emu, const float* __restrict__ esg,
                              float* __restrict__ mu_out, float* __restrict__ s_out,
                              int Bn, int H, int W, int C1,
                              int He, int We, int C2, int dh, int dw)
{
    int C = C1 + C2;
    long nidx = (long)blockIdx.x * blockDim.x + threadIdx.x;
    long tot = (long)Bn * H * W * C;
    if (nidx >= tot) return;
    int c = (int)(nidx % C); long r = nidx / C;
    int x = (int)(r % W); r /= W;
    int y = (int)(r % H); int b = (int)(r / H);
    float m, s;
    if (c < C1) {
        long i = (((long)b * H + y) * W + x) * C1 + c;
        m = dmu[i]; s = dsg[i];
    } else {
        long i = (((long)b * He + (y + dh)) * We + (x + dw)) * C2 + (c - C1);
        m = emu[i]; s = esg[i];
    }
    mu_out[nidx] = m;
    s_out[nidx]  = s;
}

// ---------------------------------------------------------------------------
// Final 1x1 conv_inter (Cin=64, Cout=2) + 2-class softmax density propagation.
// ---------------------------------------------------------------------------
__global__ void final_kernel(const float* __restrict__ mu_in,
                             const float* __restrict__ s_in,
                             const float* __restrict__ w,
                             const float* __restrict__ wsig,
                             float* __restrict__ out,
                             int Bn, int H, int W, int Cin)
{
    long nidx = (long)blockIdx.x * blockDim.x + threadIdx.x;
    long tot = (long)Bn * H * W;
    if (nidx >= tot) return;
    const float* xm = mu_in + nidx * Cin;
    const float* xs = s_in  + nidx * Cin;
    float m0 = 0.f, m1 = 0.f, s0 = 0.f, s1 = 0.f, ps = 0.f;
    for (int c = 0; c < Cin; ++c) {
        float a = xm[c], b = xs[c];
        float w0 = w[c * 2 + 0], w1 = w[c * 2 + 1];
        m0 += a * w0; m1 += a * w1;
        ps += a * a + b;
        s0 += b * w0 * w0; s1 += b * w1 * w1;
    }
    s0 += ps * softplus_f(wsig[0]);
    s1 += ps * softplus_f(wsig[1]);
    float mx = fmaxf(m0, m1);
    float e0 = expf(m0 - mx), e1 = expf(m1 - mx);
    float inv = 1.0f / (e0 + e1);
    float p0 = e0 * inv, p1 = e1 * inv;
    float q  = p0 * p1;
    float so = q * q * (s0 + s1);
    out[nidx * 2 + 0] = p0;
    out[nidx * 2 + 1] = p1;
    out[tot * 2 + nidx * 2 + 0] = so;
    out[tot * 2 + nidx * 2 + 1] = so;
}

// ---------------------------------------------------------------------------
extern "C" void kernel_launch(void* const* d_in, const int* in_sizes, int n_in,
                              void* d_out, int out_size, void* d_ws, size_t ws_size,
                              hipStream_t stream)
{
    const float* x = (const float*)d_in[0];
    auto Wm = [&](int i) { return (const float*)d_in[1 + 2 * i]; };
    auto Sg = [&](int i) { return (const float*)d_in[2 + 2 * i]; };
    // layer index: 0 e1a, 1 e1b, 2 e2a, 3 e2b, 4 ba, 5 bb,
    //              6 d2u, 7 d2a, 8 d2b, 9 d1u, 10 d1a, 11 d1b, 12 fin

    float* ws = (float*)d_ws;
    const size_t SZ_A  = 2ul * 4 * 127 * 127 * 128;  // ping arena
    const size_t SZ_B  = 2ul * 4 * 65 * 65 * 256;    // pong arena
    const size_t SZ_P1 = 2ul * 4 * 128 * 128 * 64;   // skip1
    const size_t SZ_P2 = 2ul * 4 * 64 * 64 * 128;    // skip2

    // f16 weight tables (w and w^2), K*Cout halfs each, per conv layer
    static const int wcnt[12] = {1728, 36864, 73728, 147456, 294912, 589824,
                                 294912, 294912, 147456, 73728, 73728, 36864};
    size_t woff[12]; size_t whalfs = 0;
    for (int i = 0; i < 12; ++i) { woff[i] = whalfs; whalfs += 2ul * wcnt[i]; }
    const size_t SZ_W = (whalfs + 1) / 2;            // in float units

    const size_t OFF_A = 0, OFF_B = OFF_A + SZ_A;
    const size_t OFF_P1 = OFF_B + SZ_B, OFF_P2 = OFF_P1 + SZ_P1;
    const size_t OFF_W = OFF_P2 + SZ_P2;
    if (ws_size < (OFF_W + SZ_W) * sizeof(float)) return;
    _Float16* wbase = (_Float16*)(ws + OFF_W);

    // one-time (per call) weight conversion
    for (int li = 0; li < 12; ++li) {
        int n = wcnt[li];
        wprep_kernel<<<(n + 255) / 256, 256, 0, stream>>>(
            Wm(li), wbase + woff[li], wbase + woff[li] + n, n);
    }

    const int B = 4;
    auto conv = [&](const float* mi, const float* si, int li,
                    float* mo, float* so,
                    int Hin, int Win, int Cin, int Cout, int pad, int relu) {
        int Ho = pad ? Hin : Hin - 2;
        int Wo = pad ? Win : Win - 2;
        int npix = B * Ho * Wo;
        dim3 g((npix + TMB - 1) / TMB, (Cout + TN - 1) / TN);
        conv_wmma_kernel<<<g, 128, 0, stream>>>(
            mi, si, wbase + woff[li], wbase + woff[li] + wcnt[li], Sg(li),
            mo, so, B, Hin, Win, Cin, Cout, Ho, Wo, pad, relu);
    };

    // encoder level 1
    size_t nE1 = (size_t)B * 128 * 128 * 64;
    float* mA = ws + OFF_A; float* sA = mA + nE1;
    conv(x, nullptr, 0, mA, sA, 128, 128, 3, 64, 1, 1);               // e1a
    float* muE1 = ws + OFF_P1; float* sE1 = muE1 + nE1;
    conv(mA, sA, 1, muE1, sE1, 128, 128, 64, 64, 1, 1);               // e1b -> skip1

    size_t nP1 = (size_t)B * 64 * 64 * 64;
    mA = ws + OFF_A; sA = mA + nP1;
    pool_kernel<<<(nP1 + 255) / 256, 256, 0, stream>>>(muE1, sE1, mA, sA, B, 128, 128, 64);

    // encoder level 2
    size_t nE2 = (size_t)B * 64 * 64 * 128;
    float* mB = ws + OFF_B; float* sB = mB + nE2;
    conv(mA, sA, 2, mB, sB, 64, 64, 64, 128, 1, 1);                   // e2a
    float* muE2 = ws + OFF_P2; float* sE2 = muE2 + nE2;
    conv(mB, sB, 3, muE2, sE2, 64, 64, 128, 128, 1, 1);               // e2b -> skip2

    size_t nP2 = (size_t)B * 32 * 32 * 128;
    mA = ws + OFF_A; sA = mA + nP2;
    pool_kernel<<<(nP2 + 255) / 256, 256, 0, stream>>>(muE2, sE2, mA, sA, B, 64, 64, 128);

    // bottleneck
    size_t nBa = (size_t)B * 32 * 32 * 256;
    mB = ws + OFF_B; sB = mB + nBa;
    conv(mA, sA, 4, mB, sB, 32, 32, 128, 256, 1, 1);                  // ba
    mA = ws + OFF_A; sA = mA + nBa;
    conv(mB, sB, 5, mA, sA, 32, 32, 256, 256, 1, 1);                  // bb

    // decoder level 2
    size_t nU2 = (size_t)B * 65 * 65 * 256;
    mB = ws + OFF_B; sB = mB + nU2;
    unpool_kernel<<<(nU2 + 255) / 256, 256, 0, stream>>>(mA, sA, mB, sB, B, 32, 32, 256);
    size_t nD2 = (size_t)B * 63 * 63 * 128;
    mA = ws + OFF_A; sA = mA + nD2;
    conv(mB, sB, 6, mA, sA, 65, 65, 256, 128, 0, 1);                  // d2u (VALID)
    size_t nC2 = (size_t)B * 63 * 63 * 256;
    mB = ws + OFF_B; sB = mB + nC2;
    concat_kernel<<<(nC2 + 255) / 256, 256, 0, stream>>>(
        mA, sA, muE2, sE2, mB, sB, B, 63, 63, 128, 64, 64, 128, 0, 0);
    mA = ws + OFF_A; sA = mA + nD2;
    conv(mB, sB, 7, mA, sA, 63, 63, 256, 128, 1, 1);                  // d2a
    mB = ws + OFF_B; sB = mB + nD2;
    conv(mA, sA, 8, mB, sB, 63, 63, 128, 128, 1, 1);                  // d2b

    // decoder level 1
    size_t nU1 = (size_t)B * 127 * 127 * 128;
    mA = ws + OFF_A; sA = mA + nU1;
    unpool_kernel<<<(nU1 + 255) / 256, 256, 0, stream>>>(mB, sB, mA, sA, B, 63, 63, 128);
    size_t nD1 = (size_t)B * 125 * 125 * 64;
    mB = ws + OFF_B; sB = mB + nD1;
    conv(mA, sA, 9, mB, sB, 127, 127, 128, 64, 0, 1);                 // d1u (VALID)
    size_t nC1 = (size_t)B * 125 * 125 * 128;
    mA = ws + OFF_A; sA = mA + nC1;
    concat_kernel<<<(nC1 + 255) / 256, 256, 0, stream>>>(
        mB, sB, muE1, sE1, mA, sA, B, 125, 125, 64, 128, 128, 64, 1, 1);
    mB = ws + OFF_B; sB = mB + nD1;
    conv(mA, sA, 10, mB, sB, 125, 125, 128, 64, 1, 1);                // d1a
    mA = ws + OFF_A; sA = mA + nD1;
    conv(mB, sB, 11, mA, sA, 125, 125, 64, 64, 1, 1);                 // d1b

    // final 1x1 + softmax density propagation -> d_out
    long npx = (long)B * 125 * 125;
    final_kernel<<<(npx + 255) / 256, 256, 0, stream>>>(
        mA, sA, Wm(12), Sg(12), (float*)d_out, B, 125, 125, 64);
}
